// CRLoss_8340826489594
// MI455X (gfx1250) — matrix operand order repaired and checked by reading, not compiled, not run on hardware
//
#include <hip/hip_runtime.h>

typedef _Float16 h8   __attribute__((ext_vector_type(8)));
typedef _Float16 h4   __attribute__((ext_vector_type(4)));
typedef _Float16 v16h __attribute__((ext_vector_type(16)));
typedef float    v8f  __attribute__((ext_vector_type(8)));

#define T_DIM 2048
#define D_DIM 128
#define B_DIM 16
#define BIGV  1000000000.0f
#define TH_SIM 0.9f
#define TH_DIFF 0.1f

union FragH { v16h v; h8 h[2]; };

// ---------------- Kernel 0: zero accumulators + output ----------------
__global__ void crl_init_kernel(float* acc, float* out) {
    int t = threadIdx.x;
    if (t < 32) acc[t] = 0.0f;
    if (t == 32) out[0] = 0.0f;
}

// ---------------- Kernel 1: L2-normalize rows, f32 -> f16 ----------------
// one wave32 per row of 128 floats; lane handles 4 contiguous elements
__global__ void crl_norm_kernel(const float* __restrict__ E,
                                _Float16* __restrict__ Xh) {
    int tid  = threadIdx.x;
    int lane = tid & 31;
    int row  = blockIdx.x * 8 + (tid >> 5);           // 8 rows per 256-thread block

    const float4* p = (const float4*)(E + (size_t)row * D_DIM);
    float4 v = p[lane];
    float ss = v.x * v.x + v.y * v.y + v.z * v.z + v.w * v.w;
#pragma unroll
    for (int m = 16; m >= 1; m >>= 1)
        ss += __shfl_xor(ss, m, 32);
    float w = sqrtf(ss);
    float s = 1.0f / fmaxf(w, 1e-6f);                 // EPS guard

    h4 hv;
    hv.x = (_Float16)(v.x * s);
    hv.y = (_Float16)(v.y * s);
    hv.z = (_Float16)(v.z * s);
    hv.w = (_Float16)(v.w * s);
    *(h4*)(Xh + (size_t)row * D_DIM + lane * 4) = hv;
}

// ---------------- Kernel 2: WMMA Gram tiles + biased min stats ----------------
// grid = (T/16 row-tiles, B). 256 threads = 8 waves; wave handles 16 column
// tiles as 8 pairs (two independent WMMA accumulation chains per pair).
__global__ __launch_bounds__(256)
void crl_sim_kernel(const _Float16* __restrict__ Xh,
                    const int* __restrict__ label,
                    float* __restrict__ acc) {
    __shared__ float red[8][16][4];   // [wave][row-in-tile][stat]

    const int tid  = threadIdx.x;
    const int wave = tid >> 5;
    const int lane = tid & 31;
    const int l15  = lane & 15;
    const int h    = lane >> 4;

    const int b  = blockIdx.y;
    const int m0 = blockIdx.x * 16;

    const _Float16* Xb = Xh + (size_t)b * T_DIM * D_DIM;
    const int*      Lb = label + (size_t)b * T_DIM;

    // ---- preload A fragments for this 16-row tile (K = h*16 + 0..15, matches B)
    FragH af[4];
    {
        const h8* pa = (const h8*)(Xb + (size_t)(m0 + l15) * D_DIM);
#pragma unroll
        for (int kk = 0; kk < 4; ++kk) {
            af[kk].h[0] = pa[kk * 4 + 2 * h];
            af[kk].h[1] = pa[kk * 4 + 2 * h + 1];
        }
    }

    // biased stats, all tracked as minima:
    //   minF = min(s + offF)  -> min over fake cols   (cfmin)
    //   minR = min(s + offR)  -> min over real cols   (crmin)
    //   negF = min(offF - s)  -> cfmax = -negF
    //   negR = min(offR - s)  -> crmax = -negR
    float minF[8], minR[8], negF[8], negR[8];
#pragma unroll
    for (int v = 0; v < 8; ++v) {
        minF[v] = BIGV; minR[v] = BIGV;
        negF[v] = BIGV; negR[v] = BIGV;
    }

    // ---- stream over this wave's 16 column tiles, two at a time
    const int ntbase = wave * 16;
    for (int j = 0; j < 16; j += 2) {
        const int col0 = (ntbase + j) * 16 + l15;
        const int col1 = col0 + 16;

        const bool f0 = (Lb[col0] == 0);
        const bool f1 = (Lb[col1] == 0);
        const float offF0 = f0 ? 0.0f : BIGV;
        const float offR0 = BIGV - offF0;
        const float offF1 = f1 ? 0.0f : BIGV;
        const float offR1 = BIGV - offF1;

        // B fragments: lane = column, K = h*16 + 0..15 (this code's order, same as A)
        const h8* pb0 = (const h8*)(Xb + (size_t)col0 * D_DIM);
        const h8* pb1 = (const h8*)(Xb + (size_t)col1 * D_DIM);

        v8f c0 = {};
        v8f c1 = {};
#pragma unroll
        for (int kk = 0; kk < 4; ++kk) {
            FragH bf0, bf1;
            bf0.h[0] = pb0[kk * 4 + 2 * h];
            bf0.h[1] = pb0[kk * 4 + 2 * h + 1];
            bf1.h[0] = pb1[kk * 4 + 2 * h];
            bf1.h[1] = pb1[kk * 4 + 2 * h + 1];
            c0 = __builtin_amdgcn_wmma_f32_16x16x32_f16(
                    false, af[kk].v, false, bf0.v, (short)0, c0, false, false);
            c1 = __builtin_amdgcn_wmma_f32_16x16x32_f16(
                    false, af[kk].v, false, bf1.v, (short)0, c1, false, false);
        }

        // c[v] = sim[m0 + h*8 + v][col]; combined biased updates:
        // adds pair into v_pk_add_f32, double-min collapses to v_min3_num_f32
#pragma unroll
        for (int v = 0; v < 8; ++v) {
            float s0 = c0[v];
            float s1 = c1[v];
            minF[v] = fminf(minF[v], fminf(s0 + offF0, s1 + offF1));
            minR[v] = fminf(minR[v], fminf(s0 + offR0, s1 + offR1));
            negF[v] = fminf(negF[v], fminf(offF0 - s0, offF1 - s1));
            negR[v] = fminf(negR[v], fminf(offR0 - s0, offR1 - s1));
        }
    }

    // ---- reduce across the 16 column-lanes of each half-wave (all mins)
#pragma unroll
    for (int m = 8; m >= 1; m >>= 1) {
#pragma unroll
        for (int v = 0; v < 8; ++v) {
            minF[v] = fminf(minF[v], __shfl_xor(minF[v], m, 32));
            minR[v] = fminf(minR[v], __shfl_xor(minR[v], m, 32));
            negF[v] = fminf(negF[v], __shfl_xor(negF[v], m, 32));
            negR[v] = fminf(negR[v], __shfl_xor(negR[v], m, 32));
        }
    }
    if (l15 == 0) {
#pragma unroll
        for (int v = 0; v < 8; ++v) {
            int r = h * 8 + v;
            red[wave][r][0] = minF[v];          // cfmin
            red[wave][r][1] = minR[v];          // crmin
            red[wave][r][2] = -negF[v];         // cfmax
            red[wave][r][3] = -negR[v];         // crmax
        }
    }
    __syncthreads();

    // ---- one thread per row: combine 8 waves, apply hinges, accumulate
    if (tid < 16) {
        const int r = tid;
        float a0 = BIGV, a1 = BIGV, a2 = -BIGV, a3 = -BIGV;
#pragma unroll
        for (int w = 0; w < 8; ++w) {
            a0 = fminf(a0, red[w][r][0]);   // min over fake cols
            a1 = fminf(a1, red[w][r][1]);   // min over real cols
            a2 = fmaxf(a2, red[w][r][2]);   // max over fake cols
            a3 = fmaxf(a3, red[w][r][3]);   // max over real cols
        }
        const int rl = Lb[m0 + r];
        if (rl == 0) {  // fake row: f2f_min hinge + f2r_max hinge
            float term = fmaxf(TH_SIM - a0, 0.0f) + fmaxf(a3 - TH_DIFF, 0.0f);
            atomicAdd(&acc[b], term);
        } else {        // real row: r2r_min hinge + r2f_max hinge
            float term = fmaxf(TH_SIM - a1, 0.0f) + fmaxf(a2 - TH_DIFF, 0.0f);
            atomicAdd(&acc[16 + b], term);
        }
    }
}

// ---------------- Kernel 3: masked means, validity, final scalar ----------------
__global__ void crl_final_kernel(const int* __restrict__ label,
                                 const float* __restrict__ acc,
                                 float* __restrict__ out) {
    int tid  = threadIdx.x;
    int b    = tid >> 5;          // 16 waves, one per batch
    int lane = tid & 31;

    const int* Lb = label + (size_t)b * T_DIM;
    int cf = 0, cr = 0;
    for (int t = lane; t < T_DIM; t += 32) {
        int l = Lb[t];
        cf += (l == 0);
        cr += (l == 1);
    }
#pragma unroll
    for (int m = 16; m >= 1; m >>= 1) {
        cf += __shfl_xor(cf, m, 32);
        cr += __shfl_xor(cr, m, 32);
    }
    if (lane == 0) {
        float sumF = acc[b];
        float sumR = acc[16 + b];
        float lossF = sumF / fmaxf((float)cf, 1.0f);   // loss_f2f + loss_f2r
        float lossR = sumR / fmaxf((float)cr, 1.0f);   // loss_r2r + loss_r2f
        float per_batch = (cf > 0 && cr > 0) ? (lossF + lossR) : 0.0f;
        atomicAdd(out, per_batch * (1.0f / (float)B_DIM));
    }
}

// ---------------- launch ----------------
extern "C" void kernel_launch(void* const* d_in, const int* in_sizes, int n_in,
                              void* d_out, int out_size, void* d_ws, size_t ws_size,
                              hipStream_t stream) {
    (void)in_sizes; (void)n_in; (void)out_size; (void)ws_size;

    const float* E   = (const float*)d_in[0];
    const int*   lab = (const int*)d_in[1];
    float*       out = (float*)d_out;

    float*    acc = (float*)d_ws;                         // 32 floats
    _Float16* Xh  = (_Float16*)((char*)d_ws + 256);       // B*T*D halves = 8 MB

    crl_init_kernel<<<1, 64, 0, stream>>>(acc, out);

    // B*T = 32768 rows, 8 rows per 256-thread block
    crl_norm_kernel<<<dim3((B_DIM * T_DIM) / 8), 256, 0, stream>>>(E, Xh);

    // 128 row-tiles x 16 batches
    crl_sim_kernel<<<dim3(T_DIM / 16, B_DIM), 256, 0, stream>>>(Xh, lab, acc);

    crl_final_kernel<<<1, 512, 0, stream>>>(lab, acc, out);
}
